// SimpleMambaLMHeadModel_2207613190165
// MI455X (gfx1250) — compile-verified, matrix operand order checked by reading, not run on hardware
//
#include <hip/hip_runtime.h>
#include <hip/hip_bf16.h>

// ---------------- model constants (match reference) ----------------
#define DM    1024
#define NL    4
#define DS    16
#define DC    4
#define DI    2048          // 2*DM
#define DTR   64            // DM/16
#define SEQ_L 1024
#define NXZ   (2*DI)        // 4096
#define NDBL  (DTR + 2*DS)  // 96
#define VOCAB 50277

typedef unsigned short u16;

// ---------------- WMMA / async types ----------------
typedef __bf16 v16bf __attribute__((ext_vector_type(16)));
typedef float  v8f   __attribute__((ext_vector_type(8)));
typedef int    v4i   __attribute__((ext_vector_type(4)));

#if __has_builtin(__builtin_amdgcn_global_load_async_to_lds_b128)
#define USE_ASYNC_LDS 1
typedef __attribute__((address_space(1))) v4i gv4i;
typedef __attribute__((address_space(3))) v4i lv4i;
#else
#define USE_ASYNC_LDS 0
#endif

__device__ __forceinline__ u16 f2bf(float f) {
  union { float f; unsigned u; } c; c.f = f;
  unsigned u = c.u;
  u += 0x7FFFu + ((u >> 16) & 1u);   // round-to-nearest-even
  return (u16)(u >> 16);
}

__device__ __forceinline__ int imin(int a, int b) { return a < b ? a : b; }

// =====================================================================
// GEMM (primary): C[M,N] = A[M,K] @ B[N,K]^T, A/B bf16, C f32.
// Block = 256 threads = 8 waves arranged 2(M) x 4(N); wave tile 64x32
// -> block tile 128x128.  8 x v_wmma_f32_16x16x32_bf16 per K-step of 32.
// Out-of-range rows are CLAMPED (not masked): OOB B-rows only feed
// accumulator columns that the guarded store never writes, so the inner
// loop carries no exec-mask branches or zero-fills.  K-loop is 2-stage
// double buffered so global_load_b128s overlap the WMMA batch.
// Fragment layout per ISA: lanes 0-15 hold K = [0,8)+[16,24) of the row,
// lanes 16-31 hold K = [8,16)+[24,32); row = tile_row + (lane&15).
// =====================================================================
union FragBF { v16bf v; uint4 q[2]; };

__global__ __launch_bounds__(256)
void gemm_bf16(const u16* __restrict__ A, const u16* __restrict__ B,
               float* __restrict__ C,
               int M, int N, int K, int lda, int ldb, int ldc) {
  const int lane = threadIdx.x & 31;
  const int wave = threadIdx.x >> 5;       // 0..7
  const int wm   = wave >> 2;              // 0..1  (M direction)
  const int wn   = wave & 3;               // 0..3  (N direction)
  const int m0   = blockIdx.y * 128 + wm * 64;
  const int n0   = blockIdx.x * 128 + wn * 32;
  const int ln   = lane & 15;
  const int klo  = (lane >> 4) << 3;

  // clamped per-fragment row base pointers
  const u16* pA[4];
  const u16* pB[2];
#pragma unroll
  for (int i = 0; i < 4; ++i)
    pA[i] = A + (size_t)imin(m0 + 16 * i + ln, M - 1) * lda + klo;
#pragma unroll
  for (int j = 0; j < 2; ++j)
    pB[j] = B + (size_t)imin(n0 + 16 * j + ln, N - 1) * ldb + klo;

  v8f acc[4][2] = {};
  v16bf a0[4], b0[2], a1[4], b1[2];

  auto loadset = [&](v16bf (&a)[4], v16bf (&b)[2], int kk) {
#pragma unroll
    for (int i = 0; i < 4; ++i) {
      FragBF f;
      f.q[0] = *(const uint4*)(pA[i] + kk);
      f.q[1] = *(const uint4*)(pA[i] + kk + 16);
      a[i] = f.v;
    }
#pragma unroll
    for (int j = 0; j < 2; ++j) {
      FragBF f;
      f.q[0] = *(const uint4*)(pB[j] + kk);
      f.q[1] = *(const uint4*)(pB[j] + kk + 16);
      b[j] = f.v;
    }
  };
  auto mmaset = [&](v16bf (&a)[4], v16bf (&b)[2]) {
#pragma unroll
    for (int i = 0; i < 4; ++i)
#pragma unroll
      for (int j = 0; j < 2; ++j)
        acc[i][j] = __builtin_amdgcn_wmma_f32_16x16x32_bf16(
            false, a[i], false, b[j], (short)0, acc[i][j], false, false);
  };

  loadset(a0, b0, 0);
  int k0 = 0;
  for (; k0 + 64 <= K; k0 += 64) {
    loadset(a1, b1, k0 + 32);
    mmaset(a0, b0);
    if (k0 + 64 < K) loadset(a0, b0, k0 + 64);
    mmaset(a1, b1);
  }
  if (k0 < K) mmaset(a0, b0);   // K % 64 == 32 tail

  // C/D layout: VGPR v -> row v (lanes 0-15) / row v+8 (lanes 16-31)
  const int rofs = klo;
  const int c0   = lane & 15;
#pragma unroll
  for (int i = 0; i < 4; ++i) {
#pragma unroll
    for (int j = 0; j < 2; ++j) {
      int col = n0 + 16 * j + c0;
      if (col >= N) continue;
#pragma unroll
      for (int v = 0; v < 8; ++v) {
        int r = m0 + 16 * i + v + rofs;
        if (r < M) C[(size_t)r * ldc + col] = acc[i][j][v];
      }
    }
  }
}

// =====================================================================
// GEMM (fallback, small-ws): f32 inputs converted in-register to bf16.
// Block tile 128x64, wave tile 32x32.
// =====================================================================
union BF16Frag { v16bf v; u16 u[16]; };

__device__ __forceinline__ v16bf load_frag_f32_as_bf16(
    const float* __restrict__ base, int row, int row_limit,
    int ld, int k0, int lane) {
  BF16Frag r;
  const float* p = base + (size_t)imin(row, row_limit - 1) * ld + k0 +
                   ((lane >> 4) << 3);
#pragma unroll
  for (int i = 0; i < 8; ++i) r.u[i]     = f2bf(p[i]);
#pragma unroll
  for (int i = 0; i < 8; ++i) r.u[8 + i] = f2bf(p[16 + i]);
  return r.v;
}

__global__ __launch_bounds__(256)
void gemm_f32cvt(const float* __restrict__ A, const float* __restrict__ B,
                 float* __restrict__ C,
                 int M, int N, int K, int lda, int ldb, int ldc) {
  const int lane = threadIdx.x & 31;
  const int wave = threadIdx.x >> 5;
  const int wm   = wave >> 1;
  const int wn   = wave & 1;
  const int m0   = blockIdx.y * 128 + wm * 32;
  const int n0   = blockIdx.x * 64  + wn * 32;
  const int ln   = lane & 15;

  v8f acc00 = {}, acc01 = {}, acc10 = {}, acc11 = {};
  for (int k0 = 0; k0 < K; k0 += 32) {
    v16bf a0 = load_frag_f32_as_bf16(A, m0 + ln,      M, lda, k0, lane);
    v16bf a1 = load_frag_f32_as_bf16(A, m0 + 16 + ln, M, lda, k0, lane);
    v16bf b0 = load_frag_f32_as_bf16(B, n0 + ln,      N, ldb, k0, lane);
    v16bf b1 = load_frag_f32_as_bf16(B, n0 + 16 + ln, N, ldb, k0, lane);
    acc00 = __builtin_amdgcn_wmma_f32_16x16x32_bf16(false, a0, false, b0,
                                                    (short)0, acc00, false, false);
    acc01 = __builtin_amdgcn_wmma_f32_16x16x32_bf16(false, a0, false, b1,
                                                    (short)0, acc01, false, false);
    acc10 = __builtin_amdgcn_wmma_f32_16x16x32_bf16(false, a1, false, b0,
                                                    (short)0, acc10, false, false);
    acc11 = __builtin_amdgcn_wmma_f32_16x16x32_bf16(false, a1, false, b1,
                                                    (short)0, acc11, false, false);
  }
  const int rofs = (lane >> 4) << 3;
  const int col0 = n0 + (lane & 15);
#pragma unroll
  for (int j = 0; j < 8; ++j) {
    int r0 = m0 + j + rofs;
    int r1 = m0 + 16 + j + rofs;
    if (r0 < M) {
      if (col0 < N)      C[(size_t)r0 * ldc + col0]      = acc00[j];
      if (col0 + 16 < N) C[(size_t)r0 * ldc + col0 + 16] = acc01[j];
    }
    if (r1 < M) {
      if (col0 < N)      C[(size_t)r1 * ldc + col0]      = acc10[j];
      if (col0 + 16 < N) C[(size_t)r1 * ldc + col0 + 16] = acc11[j];
    }
  }
}

// ---------------- f32 -> bf16 bulk convert (vectorized x4) ----------------
__global__ __launch_bounds__(256)
void cvt_f32_bf16_kernel(const float* __restrict__ s, u16* __restrict__ d,
                         size_t n) {
  size_t i = ((size_t)blockIdx.x * blockDim.x + threadIdx.x) * 4;
  if (i >= n) return;
  float4 v = *(const float4*)(s + i);
  uint2 o;
  o.x = (unsigned)f2bf(v.x) | ((unsigned)f2bf(v.y) << 16);
  o.y = (unsigned)f2bf(v.z) | ((unsigned)f2bf(v.w) << 16);
  *(uint2*)(d + i) = o;
}

// ---------------- embedding gather ----------------
__global__ __launch_bounds__(256)
void embed_kernel(const int* __restrict__ ids, const float* __restrict__ emb,
                  float* __restrict__ x) {
  const int t  = blockIdx.x;
  const int id = ids[t];
  const float* e = emb + (size_t)id * DM;
  float* xr = x + (size_t)t * DM;
  for (int i = threadIdx.x; i < DM; i += blockDim.x) xr[i] = e[i];
}

// ------- causal depthwise conv (DC=4) + bias + SiLU; writes f32 + bf16 ------
__global__ __launch_bounds__(256)
void conv_silu_kernel(const float* __restrict__ xz, const float* __restrict__ cw,
                      const float* __restrict__ cb, float* __restrict__ xc,
                      u16* __restrict__ xcbf) {
  const int idx = blockIdx.x * blockDim.x + threadIdx.x;
  if (idx >= SEQ_L * DI) return;
  const int d = idx & (DI - 1);
  const int t = idx / DI;
  float acc = cb[d];
#pragma unroll
  for (int k = 0; k < DC; ++k) {
    int ts = t + k - (DC - 1);
    if (ts >= 0) acc += cw[d * DC + k] * xz[(size_t)ts * NXZ + d];
  }
  float sig = 1.f / (1.f + __expf(-acc));
  float r = acc * sig;
  xc[idx]   = r;
  xcbf[idx] = f2bf(r);
}

// ---------------- dt = softplus(dt_lin + b_dt) ----------------
__global__ __launch_bounds__(256)
void bias_softplus_kernel(float* __restrict__ dt, const float* __restrict__ bdt) {
  const int idx = blockIdx.x * blockDim.x + threadIdx.x;
  if (idx >= SEQ_L * DI) return;
  const int d = idx & (DI - 1);
  float v = dt[idx] + bdt[d];
  dt[idx] = (v > 20.f) ? v : log1pf(__expf(v));
}

// ----------- selective scan; fuses D-skip, z-gating; writes f32+bf16 --------
// B_t/C_t staging uses CDNA5 async global->LDS copies when available
// (GLOBAL_LOAD_ASYNC_TO_LDS_B128 + s_wait_asynccnt), else plain LDS stores.
// B and C are adjacent columns [64,96) of dbl, so each row is one
// contiguous 128-byte transfer (8 x b128 chunks).
#define SCAN_CHUNK 128
__global__ __launch_bounds__(256)
void scan_kernel(const float* __restrict__ dt, const float* __restrict__ dbl,
                 const float* __restrict__ xc, const float* __restrict__ xz,
                 const float* __restrict__ A_log, const float* __restrict__ Dp,
                 float* __restrict__ y, u16* __restrict__ ybf) {
  const int d = blockIdx.x * blockDim.x + threadIdx.x;  // 0..DI-1
  __shared__ float sBC[SCAN_CHUNK][2 * DS];   // [t][0..15]=B, [16..31]=C

  float A[DS], h[DS];
#pragma unroll
  for (int s = 0; s < DS; ++s) {
    A[s] = -__expf(A_log[(size_t)d * DS + s]);
    h[s] = 0.f;
  }
  const float Dd = Dp[d];

  for (int t0 = 0; t0 < SEQ_L; t0 += SCAN_CHUNK) {
#if USE_ASYNC_LDS
    for (int i = threadIdx.x; i < SCAN_CHUNK * 8; i += blockDim.x) {
      const int row = i >> 3, ch = i & 7;
      const float* g = dbl + (size_t)(t0 + row) * NDBL + DTR + ch * 4;
      __builtin_amdgcn_global_load_async_to_lds_b128(
          (gv4i*)g, (lv4i*)&sBC[row][ch * 4], 0, 0);
    }
#if __has_builtin(__builtin_amdgcn_s_wait_asynccnt)
    __builtin_amdgcn_s_wait_asynccnt(0);
#else
    asm volatile("s_wait_asynccnt 0" ::: "memory");
#endif
#else
    for (int i = threadIdx.x; i < SCAN_CHUNK * 2 * DS; i += blockDim.x) {
      int tt = i >> 5, c = i & 31;
      sBC[tt][c] = dbl[(size_t)(t0 + tt) * NDBL + DTR + c];
    }
#endif
    __syncthreads();
    for (int tt = 0; tt < SCAN_CHUNK; ++tt) {
      const int t    = t0 + tt;
      const float dtv = dt[(size_t)t * DI + d];
      const float xv  = xc[(size_t)t * DI + d];
      const float dx  = dtv * xv;
      float yv = 0.f;
#pragma unroll
      for (int s = 0; s < DS; ++s) {
        float dA = __expf(dtv * A[s]);
        h[s] = h[s] * dA + dx * sBC[tt][s];
        yv  += h[s] * sBC[tt][DS + s];
      }
      const float zv  = xz[(size_t)t * NXZ + DI + d];
      const float sig = 1.f / (1.f + __expf(-zv));
      const float r   = (yv + xv * Dd) * (zv * sig);
      y[(size_t)t * DI + d]   = r;
      ybf[(size_t)t * DI + d] = f2bf(r);
    }
    __syncthreads();
  }
}

// ---------------- final LayerNorm; writes f32 + bf16 ----------------
__global__ __launch_bounds__(256)
void layernorm_kernel(const float* __restrict__ x, const float* __restrict__ g,
                      const float* __restrict__ b, float* __restrict__ o,
                      u16* __restrict__ obf) {
  const int t = blockIdx.x;
  const float* xr = x + (size_t)t * DM;
  float s = 0.f, s2 = 0.f;
  for (int i = threadIdx.x; i < DM; i += 256) {
    float v = xr[i];
    s += v; s2 += v * v;
  }
  __shared__ float rs[256], rs2[256];
  rs[threadIdx.x] = s; rs2[threadIdx.x] = s2;
  __syncthreads();
  for (int st = 128; st > 0; st >>= 1) {
    if (threadIdx.x < st) {
      rs[threadIdx.x]  += rs[threadIdx.x + st];
      rs2[threadIdx.x] += rs2[threadIdx.x + st];
    }
    __syncthreads();
  }
  const float mean = rs[0] / DM;
  const float var  = rs2[0] / DM - mean * mean;
  const float inv  = rsqrtf(var + 1e-5f);
  for (int i = threadIdx.x; i < DM; i += 256) {
    float r = (xr[i] - mean) * inv * g[i] + b[i];
    o[(size_t)t * DM + i]   = r;
    obf[(size_t)t * DM + i] = f2bf(r);
  }
}

// ---------------- host-side launch ----------------
extern "C" void kernel_launch(void* const* d_in, const int* in_sizes, int n_in,
                              void* d_out, int out_size, void* d_ws, size_t ws_size,
                              hipStream_t stream) {
  (void)in_sizes; (void)n_in; (void)out_size;

  const int*   ids    = (const int*)  d_in[0];
  const float* emb    = (const float*)d_in[1];
  const float* W_in   = (const float*)d_in[2];   // (NL, 4096, 1024)
  const float* conv_w = (const float*)d_in[3];   // (NL, 2048, 1, 4)
  const float* conv_b = (const float*)d_in[4];   // (NL, 2048)
  const float* W_x    = (const float*)d_in[5];   // (NL, 96, 2048)
  const float* W_dt   = (const float*)d_in[6];   // (NL, 2048, 64)
  const float* b_dt   = (const float*)d_in[7];   // (NL, 2048)
  const float* A_log  = (const float*)d_in[8];   // (NL, 2048, 16)
  const float* Dp     = (const float*)d_in[9];   // (NL, 2048)
  const float* W_out  = (const float*)d_in[10];  // (NL, 1024, 2048)
  const float* ln_g   = (const float*)d_in[11];
  const float* ln_b   = (const float*)d_in[12];
  const float* W_head = (const float*)d_in[13];  // (50277, 1024)
  float* out = (float*)d_out;

  // ---- workspace layout ----
  char* p = (char*)d_ws;
  auto allocF = [&](size_t n) { float* r = (float*)p; p += n * sizeof(float); return r; };
  auto allocH = [&](size_t n) { u16*   r = (u16*)p;   p += n * sizeof(u16);   return r; };

  // f32 region (always used)
  float* x   = allocF((size_t)SEQ_L * DM);
  float* xz  = allocF((size_t)SEQ_L * NXZ);
  float* xc  = allocF((size_t)SEQ_L * DI);
  float* dbl = allocF((size_t)SEQ_L * NDBL);
  float* dtb = allocF((size_t)SEQ_L * DI);
  float* yb  = allocF((size_t)SEQ_L * DI);
  float* xn  = allocF((size_t)SEQ_L * DM);
  // bf16 activation region (always used)
  u16* x_bf   = allocH((size_t)SEQ_L * DM);
  u16* xc_bf  = allocH((size_t)SEQ_L * DI);
  u16* dbl_bf = allocH((size_t)SEQ_L * NDBL);
  u16* y_bf   = allocH((size_t)SEQ_L * DI);
  u16* xn_bf  = allocH((size_t)SEQ_L * DM);
  // bf16 weight region (used only when ws is big enough)
  u16* w_in_bf  = allocH((size_t)NXZ * DM);
  u16* w_x_bf   = allocH((size_t)NDBL * DI);
  u16* w_dt_bf  = allocH((size_t)DI * DTR);
  u16* w_out_bf = allocH((size_t)DM * DI);
  u16* whead_bf = allocH((size_t)VOCAB * DM);
  const size_t needed = (size_t)(p - (char*)d_ws);
  const bool full = (ws_size >= needed);

  const dim3 blk(256);
  auto cvt = [&](const float* s, u16* d, size_t n) {
    cvt_f32_bf16_kernel<<<(unsigned)((n / 4 + 255) / 256), blk, 0, stream>>>(s, d, n);
  };
  auto gemmB = [&](const u16* A, const u16* Bw, float* C,
                   int M, int N, int K, int lda, int ldb, int ldc) {
    dim3 grid((N + 127) / 128, (M + 127) / 128);
    gemm_bf16<<<grid, blk, 0, stream>>>(A, Bw, C, M, N, K, lda, ldb, ldc);
  };
  auto gemmF = [&](const float* A, const float* Bw, float* C,
                   int M, int N, int K, int lda, int ldb, int ldc) {
    dim3 grid((N + 63) / 64, (M + 127) / 128);
    gemm_f32cvt<<<grid, blk, 0, stream>>>(A, Bw, C, M, N, K, lda, ldb, ldc);
  };

  const int nEl = SEQ_L * DI;
  embed_kernel<<<SEQ_L, blk, 0, stream>>>(ids, emb, x);
  if (full) cvt(x, x_bf, (size_t)SEQ_L * DM);

  for (int l = 0; l < NL; ++l) {
    const float* Wl_in  = W_in  + (size_t)l * NXZ * DM;
    const float* Wl_x   = W_x   + (size_t)l * NDBL * DI;
    const float* Wl_dt  = W_dt  + (size_t)l * DI * DTR;
    const float* Wl_out = W_out + (size_t)l * DM * DI;

    // xz = x @ W_in^T
    if (full) {
      cvt(Wl_in, w_in_bf, (size_t)NXZ * DM);
      gemmB(x_bf, w_in_bf, xz, SEQ_L, NXZ, DM, DM, DM, NXZ);
    } else {
      gemmF(x, Wl_in, xz, SEQ_L, NXZ, DM, DM, DM, NXZ);
    }
    // depthwise conv + SiLU
    conv_silu_kernel<<<(nEl + 255) / 256, blk, 0, stream>>>(
        xz, conv_w + (size_t)l * DI * DC, conv_b + (size_t)l * DI, xc, xc_bf);
    // dbl = xc @ W_x^T
    if (full) {
      cvt(Wl_x, w_x_bf, (size_t)NDBL * DI);
      gemmB(xc_bf, w_x_bf, dbl, SEQ_L, NDBL, DI, DI, DI, NDBL);
      cvt(dbl, dbl_bf, (size_t)SEQ_L * NDBL);
    } else {
      gemmF(xc, Wl_x, dbl, SEQ_L, NDBL, DI, DI, DI, NDBL);
    }
    // dt_lin = dbl[:, :64] @ W_dt^T  (A row stride = 96)
    if (full) {
      cvt(Wl_dt, w_dt_bf, (size_t)DI * DTR);
      gemmB(dbl_bf, w_dt_bf, dtb, SEQ_L, DI, DTR, NDBL, DTR, DI);
    } else {
      gemmF(dbl, Wl_dt, dtb, SEQ_L, DI, DTR, NDBL, DTR, DI);
    }
    bias_softplus_kernel<<<(nEl + 255) / 256, blk, 0, stream>>>(
        dtb, b_dt + (size_t)l * DI);
    // selective scan (+D skip, +z gate)
    scan_kernel<<<DI / 256, blk, 0, stream>>>(
        dtb, dbl, xc, xz, A_log + (size_t)l * DI * DS, Dp + (size_t)l * DI,
        yb, y_bf);
    // x = y @ W_out^T
    if (full) {
      cvt(Wl_out, w_out_bf, (size_t)DM * DI);
      gemmB(y_bf, w_out_bf, x, SEQ_L, DM, DI, DI, DI, DM);
      if (l + 1 < NL) cvt(x, x_bf, (size_t)SEQ_L * DM);
    } else {
      gemmF(yb, Wl_out, x, SEQ_L, DM, DI, DI, DI, DM);
    }
  }

  layernorm_kernel<<<SEQ_L, blk, 0, stream>>>(x, ln_g, ln_b, xn, xn_bf);

  // logits = xn @ W_head^T  -> d_out
  if (full) {
    cvt(W_head, whead_bf, (size_t)VOCAB * DM);
    gemmB(xn_bf, whead_bf, out, SEQ_L, VOCAB, DM, DM, DM, VOCAB);
  } else {
    gemmF(xn, W_head, out, SEQ_L, VOCAB, DM, DM, DM, VOCAB);
  }
}
